// ISTFT_75952201662485
// MI455X (gfx1250) — compile-verified, matrix operand order checked by reading, not run on hardware
//
#include <hip/hip_runtime.h>

typedef __attribute__((ext_vector_type(16))) _Float16 v16h;
typedef __attribute__((ext_vector_type(8)))  _Float16 v8h;
typedef __attribute__((ext_vector_type(2)))  _Float16 v2h;
typedef __attribute__((ext_vector_type(8)))  float    v8f;

#define N_FFT     1024
#define HOP       256
#define PADTRIM   384
#define BATCH     32
#define NBINS     513
#define T_FRAMES  2048
#define KCHUNK    1056               /* 513 cos + 513 sin, padded to 33*32 */
#define KTOT      (4 * KCHUNK)       /* 4224: 4 overlapping frames stacked */
#define NSEG      2051               /* 525056 / 256 output segments */
#define SEGTILES  129                /* ceil(2051/16): column tiles per batch */
#define OUT_PER_B 524288
#define M_ROWS    256
#define LDS_STRIDE 40                /* halves per B-tile row (16B-aligned, bank-spread) */

// ---------------------------------------------------------------------------
// Kernel 1: windowed irfft basis, f16, row-major [M_ROWS][KTOT].
// Column c = d*KCHUNK + j ; j<513 -> cos(k), 513<=j<1026 -> -sin(k), else 0.
// ---------------------------------------------------------------------------
__global__ __launch_bounds__(256) void build_basis(const float* __restrict__ window,
                                                   _Float16* __restrict__ A) {
    int idx = blockIdx.x * 256 + threadIdx.x;
    if (idx >= M_ROWS * KTOT) return;
    int r = idx / KTOT;
    int c = idx - r * KTOT;
    int d = c / KCHUNK;
    int j = c - d * KCHUNK;
    int n = d * HOP + r;                 // 0..1023
    float val = 0.0f;
    if (j < 2 * NBINS) {
        int   k  = (j < NBINS) ? j : (j - NBINS);
        float wk = (k == 0 || k == NBINS - 1) ? 1.0f : 2.0f;
        int   ph = (k * n) & (N_FFT - 1);          // exact integer phase reduction
        float ang = 6.2831853071795864769f * (float)ph * (1.0f / (float)N_FFT);
        float tr  = (j < NBINS) ? __cosf(ang) : -__sinf(ang);
        val = window[n] * wk * tr * (1.0f / (float)N_FFT);
    }
    A[idx] = (_Float16)val;
}

// ---------------------------------------------------------------------------
// Kernel 2: fused irfft-GEMM + window + overlap-add + envelope + trim.
// Grid: x = 32 batches * 129 segment tiles. Block: 256 threads = 8 waves.
// Each wave: two 16x16 output tiles (rows w*16 and 128+w*16), 2 accumulators,
// 2 x v_wmma_f32_16x16x32_f16 per 32-wide K-step, double-buffered LDS B-tile.
// ---------------------------------------------------------------------------
__global__ __launch_bounds__(256) void istft_wmma(const float* __restrict__ sr,
                                                  const float* __restrict__ si,
                                                  const float* __restrict__ window,
                                                  const _Float16* __restrict__ A,
                                                  float* __restrict__ out) {
    __shared__ _Float16 Blds[2][16 * LDS_STRIDE];

    const int tid   = threadIdx.x;
    const int wave  = tid >> 5;
    const int lane  = tid & 31;
    const int nlane = lane & 15;
    const int hsel  = (lane < 16) ? 0 : 8;     // half-wave K split per ISA layout
    const int bidx  = blockIdx.x / SEGTILES;   // batch (uniform -> SGPR)
    const int s0    = (blockIdx.x - bidx * SEGTILES) * 16;

    // Fill mapping: consecutive lanes step the column (frame) index -> coalesced.
    const int fn   = tid & 15;          // column within tile
    const int fk   = (tid >> 4) * 2;    // k-pair base: 0,2,...,30
    const int scol = s0 + fn;           // segment for this fill thread

    const _Float16* Arow0 = A + (size_t)(wave * 16 + nlane) * KTOT;
    const _Float16* Arow1 = A + (size_t)(128 + wave * 16 + nlane) * KTOT;

    v8f acc0 = {}, acc1 = {};

    auto fillTile = [&](int bf, int kkv) {
        int d  = kkv / KCHUNK;                       // uniform -> SALU
        int jj = kkv - d * KCHUNK;
        int frame   = scol - d;
        int frame_c = frame < 0 ? 0 : (frame > T_FRAMES - 1 ? T_FRAMES - 1 : frame);
        bool fok    = (frame >= 0) && (frame < T_FRAMES) && (scol < NSEG);
        v2h pv;
        #pragma unroll
        for (int kq = 0; kq < 2; ++kq) {
            int  j    = jj + fk + kq;
            bool isim = j >= NBINS;
            int  jc   = isim ? j - NBINS : j;
            int  jcc  = jc > NBINS - 1 ? NBINS - 1 : jc;   // clamp pad lanes
            const float* p = (isim ? si : sr) +
                             ((size_t)bidx * NBINS + jcc) * T_FRAMES + frame_c;
            float raw = *p;                                 // unpredicated, clamped
            float v   = (fok && j < 2 * NBINS) ? raw : 0.0f;
            pv[kq] = (_Float16)v;
        }
        *(v2h*)&Blds[bf][fn * LDS_STRIDE + fk] = pv;        // one packed store
    };

    auto loadA = [&](const _Float16* p) {
        v16h a;
        *((v8h*)&a)       = *(const v8h*)(p + hsel);
        *(((v8h*)&a) + 1) = *(const v8h*)(p + 16 + hsel);
        return a;
    };

    fillTile(0, 0);
    int buf = 0;
    for (int kk = 0; kk < KTOT; kk += 32) {
        __syncthreads();
        v16h bm;
        *((v8h*)&bm)       = *(const v8h*)(&Blds[buf][nlane * LDS_STRIDE + hsel]);
        *(((v8h*)&bm) + 1) = *(const v8h*)(&Blds[buf][nlane * LDS_STRIDE + 16 + hsel]);
        v16h a0 = loadA(Arow0 + kk);
        v16h a1 = loadA(Arow1 + kk);
        if (kk + 32 < KTOT) fillTile(buf ^ 1, kk + 32);     // overlap next gather
        acc0 = __builtin_amdgcn_wmma_f32_16x16x32_f16(false, a0, false, bm,
                                                      (short)0, acc0, false, false);
        acc1 = __builtin_amdgcn_wmma_f32_16x16x32_f16(false, a1, false, bm,
                                                      (short)0, acc1, false, false);
        buf ^= 1;
    }

    // Epilogue: C/D layout -> lanes 0-15: M=v, lanes 16-31: M=v+8; N = lane%16.
    int s = s0 + nlane;
    if (s >= NSEG) return;
    auto storeAcc = [&](const v8f& acc, int rbase) {
        #pragma unroll
        for (int v = 0; v < 8; ++v) {
            int  r      = rbase + v;
            long sample = (long)s * HOP + r;
            long o      = sample - PADTRIM;
            if (o >= 0 && o < OUT_PER_B) {
                float env = 0.0f;
                #pragma unroll
                for (int d = 0; d < 4; ++d) {
                    int fr = s - d;
                    if (fr >= 0 && fr < T_FRAMES) {
                        float w = window[d * HOP + r];
                        env += w * w;
                    }
                }
                out[(size_t)bidx * OUT_PER_B + o] = acc[v] / env;
            }
        }
    };
    int rb = wave * 16 + ((lane < 16) ? 0 : 8);
    storeAcc(acc0, rb);
    storeAcc(acc1, 128 + rb);
}

// ---------------------------------------------------------------------------
extern "C" void kernel_launch(void* const* d_in, const int* in_sizes, int n_in,
                              void* d_out, int out_size, void* d_ws, size_t ws_size,
                              hipStream_t stream) {
    (void)in_sizes; (void)n_in; (void)out_size; (void)ws_size;
    const float* sr  = (const float*)d_in[0];
    const float* si  = (const float*)d_in[1];
    const float* win = (const float*)d_in[2];
    float* out = (float*)d_out;
    _Float16* A = (_Float16*)d_ws;   // 256*4224*2 B = 2.1 MB basis (L2-resident)

    int totalA = M_ROWS * KTOT;
    build_basis<<<(totalA + 255) / 256, 256, 0, stream>>>(win, A);

    istft_wmma<<<BATCH * SEGTILES, 256, 0, stream>>>(sr, si, win, A, out);
}
